// DecisionTransformer_85255100826051
// MI455X (gfx1250) — compile-verified, matrix operand order checked by reading, not run on hardware
//
#include <hip/hip_runtime.h>
#include <math.h>

// ---------------------------------------------------------------------------
// Decision Transformer forward for MI455X (gfx1250, wave32, WMMA).
// All GEMM operands f16 in memory -> pure b128 loads feeding
// v_wmma_f32_16x16x32_f16 (4 WMMA / K-step, 16x64 wave tile), with a
// ping-pong double buffer so loads of tile k+2 overlap WMMAs of tile k+1
// without register copies. TreeLSTM restructured into batched GEMMs +
// per-tree LDS scatter kernels.
// ---------------------------------------------------------------------------

typedef __attribute__((ext_vector_type(16))) _Float16 v16h;
typedef __attribute__((ext_vector_type(8)))  _Float16 v8h;
typedef __attribute__((ext_vector_type(4)))  _Float16 v4h;
typedef __attribute__((ext_vector_type(8)))  float    v8f;

namespace cfg {
constexpr int B = 16, T = 32, NA = 16, NN = 31, E = 30, ND = 12, AA = 83;
constexpr int H = 128, HS = 128, SD = 256, AD = 5, D = 128, NH = 4, NL = 3;
constexpr int DFF = 2048, LEVELS = 5;
constexpr int NT   = B * T * NA;   // 8192 trees
constexpr int R    = NT * NN;      // 253952 node rows
constexpr int RE   = NT * E;       // 245760 edge rows
constexpr int BN   = B * NA;       // 256 sequences
constexpr int S3   = 3 * T;        // 96 tokens per sequence
constexpr int ROWS = BN * S3;      // 24576 token rows
constexpr int AAP  = 96;           // AA=83 zero-padded to 96 (K % 32 == 0)
constexpr float EPS = 1e-5f;
}

__device__ __forceinline__ float sigf(float x) { return 1.f / (1.f + __expf(-x)); }
__device__ __forceinline__ float geluf(float x) {
  return 0.5f * x * (1.f + erff(x * 0.70710678118654752f));
}
__device__ __forceinline__ float wred_sum(float v) {
#pragma unroll
  for (int off = 16; off; off >>= 1) v += __shfl_xor(v, off, 32);
  return v;
}
__device__ __forceinline__ float wred_max(float v) {
#pragma unroll
  for (int off = 16; off; off >>= 1) v = fmaxf(v, __shfl_xor(v, off, 32));
  return v;
}
__device__ __forceinline__ v16h catv(v8h lo, v8h hi) {
  return __builtin_shufflevector(lo, hi, 0,1,2,3,4,5,6,7,8,9,10,11,12,13,14,15);
}

// One K-step's operands for a 16x64 wave tile: A + 4 B tiles, 10 b128 loads.
struct OpsH { v8h aL, aH, b0L, b0H, b1L, b1H, b2L, b2H, b3L, b3H; };

__device__ __forceinline__ OpsH load_ops(const _Float16* __restrict__ Ar,
                                         const _Float16* __restrict__ W0,
                                         const _Float16* __restrict__ W1,
                                         const _Float16* __restrict__ W2,
                                         const _Float16* __restrict__ W3,
                                         int kA, int kB) {
  OpsH o;
  o.aL  = *(const v8h*)(Ar + kA); o.aH  = *(const v8h*)(Ar + kB);
  o.b0L = *(const v8h*)(W0 + kA); o.b0H = *(const v8h*)(W0 + kB);
  o.b1L = *(const v8h*)(W1 + kA); o.b1H = *(const v8h*)(W1 + kB);
  o.b2L = *(const v8h*)(W2 + kA); o.b2H = *(const v8h*)(W2 + kB);
  o.b3L = *(const v8h*)(W3 + kA); o.b3H = *(const v8h*)(W3 + kB);
  return o;
}

// ---------------------------------------------------------------------------
// WMMA GEMM: C[M,N] = act(A[M,K] @ W[N,K]^T + bias), A/W f16, accum f32.
// Requirements (all call sites satisfy): M%64==0, N%128==0, K%32==0 (K>=96),
// 16-byte aligned rows. Block = 8 waves, 4(M) x 2(N); wave tile 16x64.
// act: 0=none, 1=relu, 2=gelu. store16: write C as f16 (feeds next GEMM).
// Operand layout per CDNA5 ISA 7.12.2: lanes<16 K 0-7/16-23, lanes>=16
// K 8-15/24-31 -> two contiguous b128 loads per operand per K-step.
// ---------------------------------------------------------------------------
__global__ void gemm_f16_wmma(const _Float16* __restrict__ A, int lda,
                              const _Float16* __restrict__ W,
                              const float* __restrict__ bias,
                              void* __restrict__ Cv, int ldc, int ccol,
                              int K, int act, int store16)
{
  const int wid  = threadIdx.x >> 5;
  const int lane = threadIdx.x & 31;
  const int lid  = lane & 15;
  const int half = lane >> 4;
  const int m0  = blockIdx.x * 64 + (wid & 3) * 16;
  const int wn0 = blockIdx.y * 128 + (wid >> 2) * 64;

  const _Float16* Arow  = A + (size_t)(m0 + lid) * lda;
  const _Float16* Wrow0 = W + (size_t)(wn0 + 0  + lid) * K;
  const _Float16* Wrow1 = W + (size_t)(wn0 + 16 + lid) * K;
  const _Float16* Wrow2 = W + (size_t)(wn0 + 32 + lid) * K;
  const _Float16* Wrow3 = W + (size_t)(wn0 + 48 + lid) * K;

  const int kAo = half * 8;
  const int kBo = 16 + half * 8;

  v8f acc[4] = {{}, {}, {}, {}};
  auto domm = [&](const OpsH& o) {
    const v16h av = catv(o.aL, o.aH);
    const v16h b0 = catv(o.b0L, o.b0H);
    const v16h b1 = catv(o.b1L, o.b1H);
    const v16h b2 = catv(o.b2L, o.b2H);
    const v16h b3 = catv(o.b3L, o.b3H);
    acc[0] = __builtin_amdgcn_wmma_f32_16x16x32_f16(false, av, false, b0, (short)0, acc[0], false, false);
    acc[1] = __builtin_amdgcn_wmma_f32_16x16x32_f16(false, av, false, b1, (short)0, acc[1], false, false);
    acc[2] = __builtin_amdgcn_wmma_f32_16x16x32_f16(false, av, false, b2, (short)0, acc[2], false, false);
    acc[3] = __builtin_amdgcn_wmma_f32_16x16x32_f16(false, av, false, b3, (short)0, acc[3], false, false);
  };

  // Prime the prefetcher for the tiles after the first two (one-shot, no
  // per-iteration branches) and load tiles 0 and 1 into the ping-pong pair.
  __builtin_prefetch(Arow + 64, 0, 3);       // global_prefetch_b8
  __builtin_prefetch(Wrow0 + 64, 0, 3);
  OpsH bufA = load_ops(Arow, Wrow0, Wrow1, Wrow2, Wrow3, kAo, kBo);
  OpsH bufB = load_ops(Arow, Wrow0, Wrow1, Wrow2, Wrow3, 32 + kAo, 32 + kBo);

  // Ping-pong: at loop top bufA holds tile k0, bufB holds tile k0+32.
  int k0 = 0;
  for (;;) {
    domm(bufA);                               // tile k0
    k0 += 32;
    if (k0 >= K) break;
    if (k0 + 32 < K)                          // preload tile k0+32 -> bufA
      bufA = load_ops(Arow, Wrow0, Wrow1, Wrow2, Wrow3, k0 + 32 + kAo, k0 + 32 + kBo);
    domm(bufB);                               // tile k0
    k0 += 32;
    if (k0 >= K) break;
    if (k0 + 32 < K)                          // preload tile k0+32 -> bufB
      bufB = load_ops(Arow, Wrow0, Wrow1, Wrow2, Wrow3, k0 + 32 + kAo, k0 + 32 + kBo);
  }

  float*    C32 = (float*)Cv;
  _Float16* C16 = (_Float16*)Cv;
#pragma unroll
  for (int nt = 0; nt < 4; ++nt) {
    const int n = wn0 + nt * 16 + lid;       // D layout: lane&15 = column
    const float bb = bias ? bias[n] : 0.f;
#pragma unroll
    for (int i = 0; i < 8; ++i) {
      const int m = m0 + i + half * 8;       // VGPR i: rows 0-7 / 8-15
      float v = acc[nt][i] + bb;
      if (act == 1)      v = fmaxf(v, 0.f);
      else if (act == 2) v = geluf(v);
      if (store16) C16[(size_t)m * ldc + ccol + n] = (_Float16)v;
      else         C32[(size_t)m * ldc + ccol + n] = v;
    }
  }
}

// f32 -> f16, 4 elements per thread (b128 in, b64 out).
__global__ void cvt_f16_4(const float* __restrict__ src, _Float16* __restrict__ dst, int n4)
{
  const int i = blockIdx.x * blockDim.x + threadIdx.x;
  if (i >= n4) return;
  const float4 f = ((const float4*)src)[i];
  v4h o;
  o[0] = (_Float16)f.x; o[1] = (_Float16)f.y; o[2] = (_Float16)f.z; o[3] = (_Float16)f.w;
  ((v4h*)dst)[i] = o;
}

// Zero-pad rows into f16: dst[r,0:sk]=src[r,0:sk]; dst[r,sk:dk]=0.
__global__ void pad_rows_f16(const float* __restrict__ src, _Float16* __restrict__ dst,
                             int rows, int sk, int dk)
{
  size_t idx = (size_t)blockIdx.x * blockDim.x + threadIdx.x;
  if (idx >= (size_t)rows * dk) return;
  const int r = (int)(idx / dk), c = (int)(idx % dk);
  dst[idx] = (c < sk) ? (_Float16)src[(size_t)r * sk + c] : (_Float16)0.f;
}

// ---------------------------------------------------------------------------
// TreeLSTM kernels (iou_x / f_x K=12 folded as VALU dots).
// ---------------------------------------------------------------------------
__global__ void tree_level0(const float* __restrict__ fx, const int* __restrict__ norder,
                            const float* __restrict__ W_iou, const float* __restrict__ b_iou,
                            float* __restrict__ h, float* __restrict__ c)
{
  size_t idx = (size_t)blockIdx.x * blockDim.x + threadIdx.x;
  if (idx >= (size_t)cfg::R * 128) return;
  const int r = (int)(idx >> 7), d = (int)(idx & 127);
  const float* xr = fx + (size_t)r * cfg::ND;
  float i = b_iou[d], o = b_iou[128 + d], u = b_iou[256 + d];
#pragma unroll
  for (int k = 0; k < cfg::ND; ++k) {
    const float xv = xr[k];
    i += xv * W_iou[(size_t)d * cfg::ND + k];
    o += xv * W_iou[(size_t)(128 + d) * cfg::ND + k];
    u += xv * W_iou[(size_t)(256 + d) * cfg::ND + k];
  }
  const float cn = sigf(i) * tanhf(u);
  const float hn = sigf(o) * tanhf(cn);
  const bool m = (norder[r] == 0);
  h[idx] = m ? hn : 0.f;
  c[idx] = m ? cn : 0.f;
}

// Gather children h rows, emit f16 for the U_f GEMM.
__global__ void tree_gather_ch(const float* __restrict__ h, const int* __restrict__ adj,
                               _Float16* __restrict__ ch)
{
  size_t idx = (size_t)blockIdx.x * blockDim.x + threadIdx.x;
  if (idx >= (size_t)cfg::RE * 128) return;
  const int d  = (int)(idx & 127);
  const int er = (int)(idx >> 7);            // tree*E + e
  const int tree = er / cfg::E;
  const int child = adj[(size_t)er * 2 + 1];
  ch[idx] = (_Float16)h[((size_t)tree * cfg::NN + child) * 128 + d];
}

// One block per tree; thread d owns feature column d -> no atomics/barriers.
// Accumulates in fp32 LDS; emits h_sum as f16 (feeds U_iou GEMM), c_sum fp32.
// Dynamic LDS: 2 * NN * 128 floats (~31 KB, fits 320 KB/WGP).
__global__ void tree_scatter(const _Float16* __restrict__ ch, const float* __restrict__ fu,
                             const float* __restrict__ fx,
                             const float* __restrict__ W_f, const float* __restrict__ b_f,
                             const int* __restrict__ adj, const int* __restrict__ eorder,
                             const float* __restrict__ c,
                             _Float16* __restrict__ h_sum, float* __restrict__ c_sum, int lvl)
{
  extern __shared__ float sm[];
  float* hs = sm;
  float* cs = sm + cfg::NN * 128;
  const int tree = blockIdx.x, d = threadIdx.x;
  for (int nd = 0; nd < cfg::NN; ++nd) { hs[nd * 128 + d] = 0.f; cs[nd * 128 + d] = 0.f; }
  float wf[cfg::ND];
#pragma unroll
  for (int k = 0; k < cfg::ND; ++k) wf[k] = W_f[(size_t)d * cfg::ND + k];
  const float bfv = b_f[d];
  const int* adjt = adj    + (size_t)tree * cfg::E * 2;
  const int* eot  = eorder + (size_t)tree * cfg::E;
  for (int e = 0; e < cfg::E; ++e) {
    const int p = adjt[2 * e], ci = adjt[2 * e + 1];
    const float em = (eot[e] == lvl) ? 1.f : 0.f;
    const size_t erow = ((size_t)tree * cfg::E + e) * 128 + d;
    const float chv = (float)ch[erow];
    const float* xr = fx + ((size_t)tree * cfg::NN + p) * cfg::ND;
    float fxv = bfv;
#pragma unroll
    for (int k = 0; k < cfg::ND; ++k) fxv += xr[k] * wf[k];
    const float f = sigf(fxv + fu[erow]);
    hs[p * 128 + d] += chv * em;
    cs[p * 128 + d] += f * c[((size_t)tree * cfg::NN + ci) * 128 + d] * em;
  }
  for (int nd = 0; nd < cfg::NN; ++nd) {
    const size_t o = ((size_t)tree * cfg::NN + nd) * 128 + d;
    h_sum[o] = (_Float16)hs[nd * 128 + d];
    c_sum[o] = cs[nd * 128 + d];
  }
}

__global__ void tree_update(const float* __restrict__ fx, const float* __restrict__ du,
                            const float* __restrict__ c_sum, const int* __restrict__ norder,
                            const float* __restrict__ W_iou, const float* __restrict__ b_iou,
                            float* __restrict__ h, float* __restrict__ c, int lvl)
{
  size_t idx = (size_t)blockIdx.x * blockDim.x + threadIdx.x;
  if (idx >= (size_t)cfg::R * 128) return;
  const int r = (int)(idx >> 7), d = (int)(idx & 127);
  const float* xr = fx + (size_t)r * cfg::ND;
  const float* dur = du + (size_t)r * 384;
  float i = b_iou[d] + dur[d];
  float o = b_iou[128 + d] + dur[128 + d];
  float u = b_iou[256 + d] + dur[256 + d];
#pragma unroll
  for (int k = 0; k < cfg::ND; ++k) {
    const float xv = xr[k];
    i += xv * W_iou[(size_t)d * cfg::ND + k];
    o += xv * W_iou[(size_t)(128 + d) * cfg::ND + k];
    u += xv * W_iou[(size_t)(256 + d) * cfg::ND + k];
  }
  const float cn = sigf(i) * tanhf(u) + c_sum[idx];
  const float hn = sigf(o) * tanhf(cn);
  if (norder[r] == lvl) { h[idx] = hn; c[idx] = cn; }
}

// Root h -> f16 cat buffer (cols 0..127 of the concat input to proj).
__global__ void root_gather(const float* __restrict__ h, _Float16* __restrict__ cat)
{
  size_t idx = (size_t)blockIdx.x * blockDim.x + threadIdx.x;
  if (idx >= (size_t)cfg::NT * 128) return;
  const int tree = (int)(idx >> 7), d = (int)(idx & 127);
  cat[(size_t)tree * 256 + d] = (_Float16)h[((size_t)tree * cfg::NN) * 128 + d];
}

// ---------------------------------------------------------------------------
// Token assembly: reproduces stack(axis=3).reshape(B,T*3,N,D) then
// transpose(0,2,1,3).reshape(B*N,96,D) index arithmetic exactly.
// ---------------------------------------------------------------------------
__global__ void build_x_kernel(const float* __restrict__ s, const float* __restrict__ rtgs,
                               const int* __restrict__ actions,
                               const float* __restrict__ er_w, const float* __restrict__ er_b,
                               const float* __restrict__ ea, float* __restrict__ x)
{
  size_t idx = (size_t)blockIdx.x * blockDim.x + threadIdx.x;
  if (idx >= (size_t)cfg::ROWS * 128) return;
  const int d  = (int)(idx & 127);
  const int u  = (int)((idx >> 7) % cfg::S3);
  const int bn = (int)(idx / ((size_t)cfg::S3 * 128));
  const int b = bn >> 4, n = bn & 15;
  const int Fp  = (u * cfg::NA + n) * 128 + d;
  const int t   = Fp / (3 * cfg::NA * 128);
  const int rem = Fp % (3 * cfg::NA * 128);
  const int n0  = rem / 384;
  const int kk  = (rem % 384) >> 7;
  const int d0  = rem & 127;
  const int base = (b * cfg::T + t) * cfg::NA + n0;
  float val;
  if (kk == 0)      val = rtgs[base] * er_w[d0] + er_b[d0];
  else if (kk == 1) val = s[(size_t)base * 128 + d0];
  else              val = ea[(size_t)actions[base] * 128 + d0];
  x[idx] = val;
}

// ---------------------------------------------------------------------------
// Attention: one block per (sequence, head); one wave per query row; shuffle
// reductions; K transposed in LDS. Output f16 (feeds out-proj GEMM).
// ---------------------------------------------------------------------------
__global__ void attention_kernel(const float* __restrict__ qkv, _Float16* __restrict__ out)
{
  __shared__ float q[96 * 32], kT[32 * 96], v[96 * 32];
  __shared__ float att[4][96];
  const int bh = blockIdx.x, bn = bh >> 2, hd = bh & 3;
  const int tid = threadIdx.x;
  for (int i = tid; i < 96 * 32; i += 128) {
    const int t = i >> 5, d = i & 31;
    const float* row = qkv + ((size_t)bn * 96 + t) * 384 + hd * 32 + d;
    q[i] = row[0];
    kT[d * 96 + t] = row[128];
    v[i] = row[256];
  }
  __syncthreads();
  const int wave = tid >> 5, lane = tid & 31;
  const float scale = 0.17677669529663687f;   // 1/sqrt(32)
  for (int i = wave; i < 96; i += 4) {
    float s[3];
#pragma unroll
    for (int cblk = 0; cblk < 3; ++cblk) {
      const int j = lane + 32 * cblk;
      float acc = 0.f;
#pragma unroll 8
      for (int x2 = 0; x2 < 32; ++x2) acc += q[i * 32 + x2] * kT[x2 * 96 + j];
      s[cblk] = acc * scale;
    }
    const float mx = wred_max(fmaxf(s[0], fmaxf(s[1], s[2])));
    const float e0 = __expf(s[0] - mx), e1 = __expf(s[1] - mx), e2 = __expf(s[2] - mx);
    const float inv = 1.f / wred_sum(e0 + e1 + e2);
    att[wave][lane] = e0 * inv; att[wave][lane + 32] = e1 * inv; att[wave][lane + 64] = e2 * inv;
    float o = 0.f;
    for (int j = 0; j < 96; ++j) o += att[wave][j] * v[j * 32 + lane];
    out[((size_t)bn * 96 + i) * 128 + hd * 32 + lane] = (_Float16)o;
  }
}

// One wave per row; no barriers.
__global__ void add_layernorm(float* __restrict__ x, const float* __restrict__ res,
                              const float* __restrict__ g, const float* __restrict__ b)
{
  const int wave = threadIdx.x >> 5, lane = threadIdx.x & 31;
  const int r = blockIdx.x * 8 + wave;
  const size_t base = (size_t)r * 128;
  float v[4];
  float sum = 0.f;
#pragma unroll
  for (int cblk = 0; cblk < 4; ++cblk) {
    const int d = lane + 32 * cblk;
    v[cblk] = x[base + d] + res[base + d];
    sum += v[cblk];
  }
  const float mu = wred_sum(sum) * (1.f / 128.f);
  float s2 = 0.f;
#pragma unroll
  for (int cblk = 0; cblk < 4; ++cblk) { v[cblk] -= mu; s2 += v[cblk] * v[cblk]; }
  const float rstd = rsqrtf(wred_sum(s2) * (1.f / 128.f) + cfg::EPS);
#pragma unroll
  for (int cblk = 0; cblk < 4; ++cblk) {
    const int d = lane + 32 * cblk;
    x[base + d] = v[cblk] * rstd * g[d] + b[d];
  }
}

__global__ void head_kernel(const float* __restrict__ x, const float* __restrict__ w1,
                            const float* __restrict__ b1, const float* __restrict__ w2,
                            const float* __restrict__ b2, float* __restrict__ out)
{
  __shared__ float last[128], t1[128];
  const int bn = blockIdx.x, d = threadIdx.x;
  last[d] = x[((size_t)bn * 96 + 95) * 128 + d];
  __syncthreads();
  float acc = b1[d];
  for (int kk = 0; kk < 128; ++kk) acc += last[kk] * w1[(size_t)d * 128 + kk];
  t1[d] = geluf(acc);
  __syncthreads();
  if (d < cfg::AD) {
    float a = b2[d];
    for (int kk = 0; kk < 128; ++kk) a += t1[kk] * w2[(size_t)d * 128 + kk];
    out[(size_t)bn * cfg::AD + d] = a;
  }
}

// ---------------------------------------------------------------------------
extern "C" void kernel_launch(void* const* d_in, const int* in_sizes, int n_in,
                              void* d_out, int out_size, void* d_ws, size_t ws_size,
                              hipStream_t stream)
{
  using namespace cfg;
  (void)in_sizes; (void)n_in; (void)out_size; (void)ws_size;

  const float* forest  = (const float*)d_in[0];
  const float* agent   = (const float*)d_in[1];
  const int*   adj     = (const int*)d_in[2];
  const int*   norder  = (const int*)d_in[3];
  const int*   eorder  = (const int*)d_in[4];
  const int*   actions = (const int*)d_in[5];
  const float* rtgs    = (const float*)d_in[6];
  // d_in[7] = timesteps (unused by the reference forward)

  // Params flattened in dict insertion order.
  const float* W_iou   = (const float*)d_in[8];
  const float* b_iou   = (const float*)d_in[9];
  const float* U_iou   = (const float*)d_in[10];
  const float* W_f     = (const float*)d_in[11];
  const float* b_f     = (const float*)d_in[12];
  const float* U_f     = (const float*)d_in[13];
  const float* attr_w1 = (const float*)d_in[14];
  const float* attr_b1 = (const float*)d_in[15];
  const float* attr_w2 = (const float*)d_in[16];
  const float* attr_b2 = (const float*)d_in[17];
  const float* proj_w  = (const float*)d_in[18];
  const float* proj_b  = (const float*)d_in[19];
  const float* es_w    = (const float*)d_in[20];
  const float* es_b    = (const float*)d_in[21];
  const float* ea      = (const float*)d_in[22];
  const float* er_w    = (const float*)d_in[23];
  const float* er_b    = (const float*)d_in[24];
  // d_in[25] = et (unused)
  const float* pa_w1   = (const float*)d_in[26];
  const float* pa_b1   = (const float*)d_in[27];
  const float* pa_w2   = (const float*)d_in[28];
  const float* pa_b2   = (const float*)d_in[29];
  struct Layer { const float *in_w,*in_b,*out_w,*out_b,*ln1_g,*ln1_b,*ln2_g,*ln2_b,*ff1_w,*ff1_b,*ff2_w,*ff2_b; } L[NL];
  for (int l = 0; l < NL; ++l) {
    const int q = 30 + 12 * l;
    L[l].in_w  = (const float*)d_in[q + 0];  L[l].in_b  = (const float*)d_in[q + 1];
    L[l].out_w = (const float*)d_in[q + 2];  L[l].out_b = (const float*)d_in[q + 3];
    L[l].ln1_g = (const float*)d_in[q + 4];  L[l].ln1_b = (const float*)d_in[q + 5];
    L[l].ln2_g = (const float*)d_in[q + 6];  L[l].ln2_b = (const float*)d_in[q + 7];
    L[l].ff1_w = (const float*)d_in[q + 8];  L[l].ff1_b = (const float*)d_in[q + 9];
    L[l].ff2_w = (const float*)d_in[q + 10]; L[l].ff2_b = (const float*)d_in[q + 11];
  }

  // ---- Workspace layout ----
  char* wsp = (char*)d_ws;
  size_t off = 0;
  auto allocB = [&](size_t bytes) -> char* {
    char* p = wsp + off;
    off = (off + bytes + 255) & ~(size_t)255;
    return p;
  };
  float*    hbuf  = (float*)allocB((size_t)R * 128 * 4);
  float*    cbuf  = (float*)allocB((size_t)R * 128 * 4);
  float*    csum  = (float*)allocB((size_t)R * 128 * 4);
  float*    fub   = (float*)allocB((size_t)RE * 128 * 4);
  _Float16* chh   = (_Float16*)allocB((size_t)RE * 128 * 2);
  _Float16* hsumh = (_Float16*)allocB((size_t)R * 128 * 2);
  _Float16* U_fh  = (_Float16*)allocB((size_t)128 * 128 * 2);
  _Float16* U_iouh= (_Float16*)allocB((size_t)384 * 128 * 2);
  char*     dubB  = allocB((size_t)R * 384 * 4);   // 390 MB; aliased below
  float*    dub   = (float*)dubB;
  size_t toff = 0;
  auto talloc = [&](size_t bytes) -> char* {
    char* p = dubB + toff;
    toff = (toff + bytes + 255) & ~(size_t)255;
    return p;
  };
  float*    sb     = (float*)talloc((size_t)NT * 128 * 4);
  float*    xb     = (float*)talloc((size_t)ROWS * 128 * 4);
  float*    qkvb   = (float*)talloc((size_t)ROWS * 384 * 4);
  float*    ob     = (float*)talloc((size_t)ROWS * 128 * 4);
  _Float16* a1h    = (_Float16*)talloc((size_t)NT * 256 * 2);
  _Float16* cath   = (_Float16*)talloc((size_t)NT * 256 * 2);
  _Float16* stateh = (_Float16*)talloc((size_t)NT * 256 * 2);
  _Float16* agentPh= (_Float16*)talloc((size_t)NT * AAP * 2);
  _Float16* w1Ph   = (_Float16*)talloc((size_t)256 * AAP * 2);
  _Float16* wtmp   = (_Float16*)talloc((size_t)DFF * 128 * 2);
  _Float16* xh     = (_Float16*)talloc((size_t)ROWS * 128 * 2);
  _Float16* atth   = (_Float16*)talloc((size_t)ROWS * 128 * 2);
  _Float16* ffhh   = (_Float16*)talloc((size_t)ROWS * DFF * 2);

  const int TPB = 256;
  auto cdiv = [](size_t a, size_t b) { return (unsigned)((a + b - 1) / b); };

  // ---- TreeLSTM ----
  cvt_f16_4<<<cdiv(128 * 128 / 4, TPB), TPB, 0, stream>>>(U_f, U_fh, 128 * 128 / 4);
  cvt_f16_4<<<cdiv(384 * 128 / 4, TPB), TPB, 0, stream>>>(U_iou, U_iouh, 384 * 128 / 4);
  tree_level0<<<cdiv((size_t)R * 128, TPB), TPB, 0, stream>>>(forest, norder, W_iou, b_iou, hbuf, cbuf);
  for (int lvl = 1; lvl < LEVELS; ++lvl) {
    tree_gather_ch<<<cdiv((size_t)RE * 128, TPB), TPB, 0, stream>>>(hbuf, adj, chh);
    gemm_f16_wmma<<<dim3(RE / 64, 1), TPB, 0, stream>>>(chh, 128, U_fh, nullptr, fub, 128, 0, 128, 0, 0);
    tree_scatter<<<NT, 128, 2 * NN * 128 * 4, stream>>>(chh, fub, forest, W_f, b_f, adj, eorder, cbuf, hsumh, csum, lvl);
    gemm_f16_wmma<<<dim3(R / 64, 3), TPB, 0, stream>>>(hsumh, 128, U_iouh, nullptr, dub, 384, 0, 128, 0, 0);
    tree_update<<<cdiv((size_t)R * 128, TPB), TPB, 0, stream>>>(forest, dub, csum, norder, W_iou, b_iou, hbuf, cbuf, lvl);
  }
  root_gather<<<cdiv((size_t)NT * 128, TPB), TPB, 0, stream>>>(hbuf, cath);

  // ---- Attribute / state embedding MLPs (K=83 zero-padded to 96) ----
  pad_rows_f16<<<cdiv((size_t)NT * AAP, TPB), TPB, 0, stream>>>(agent, agentPh, NT, AA, AAP);
  pad_rows_f16<<<cdiv((size_t)256 * AAP, TPB), TPB, 0, stream>>>(attr_w1, w1Ph, 256, AA, AAP);
  gemm_f16_wmma<<<dim3(NT / 64, 2), TPB, 0, stream>>>(agentPh, AAP, w1Ph, attr_b1, a1h, 256, 0, AAP, 2, 1);
  cvt_f16_4<<<cdiv(128 * 256 / 4, TPB), TPB, 0, stream>>>(attr_w2, wtmp, 128 * 256 / 4);
  gemm_f16_wmma<<<dim3(NT / 64, 1), TPB, 0, stream>>>(a1h, 256, wtmp, attr_b2, cath, 256, 128, 256, 2, 1);
  cvt_f16_4<<<cdiv(256 * 256 / 4, TPB), TPB, 0, stream>>>(proj_w, wtmp, 256 * 256 / 4);
  gemm_f16_wmma<<<dim3(NT / 64, 2), TPB, 0, stream>>>(cath, 256, wtmp, proj_b, stateh, 256, 0, 256, 0, 1);
  cvt_f16_4<<<cdiv(128 * 256 / 4, TPB), TPB, 0, stream>>>(es_w, wtmp, 128 * 256 / 4);
  gemm_f16_wmma<<<dim3(NT / 64, 1), TPB, 0, stream>>>(stateh, 256, wtmp, es_b, sb, 128, 0, 256, 0, 0);

  // ---- Token assembly ----
  build_x_kernel<<<cdiv((size_t)ROWS * 128, TPB), TPB, 0, stream>>>(sb, rtgs, actions, er_w, er_b, ea, xb);

  // ---- Transformer encoder x3 ----
  const int n4x = ROWS * 128 / 4;
  for (int l = 0; l < NL; ++l) {
    cvt_f16_4<<<cdiv(n4x, TPB), TPB, 0, stream>>>(xb, xh, n4x);
    cvt_f16_4<<<cdiv(384 * 128 / 4, TPB), TPB, 0, stream>>>(L[l].in_w, wtmp, 384 * 128 / 4);
    gemm_f16_wmma<<<dim3(ROWS / 64, 3), TPB, 0, stream>>>(xh, 128, wtmp, L[l].in_b, qkvb, 384, 0, 128, 0, 0);
    attention_kernel<<<BN * NH, 128, 0, stream>>>(qkvb, atth);
    cvt_f16_4<<<cdiv(128 * 128 / 4, TPB), TPB, 0, stream>>>(L[l].out_w, wtmp, 128 * 128 / 4);
    gemm_f16_wmma<<<dim3(ROWS / 64, 1), TPB, 0, stream>>>(atth, 128, wtmp, L[l].out_b, ob, 128, 0, 128, 0, 0);
    add_layernorm<<<ROWS / 8, TPB, 0, stream>>>(xb, ob, L[l].ln1_g, L[l].ln1_b);
    cvt_f16_4<<<cdiv(n4x, TPB), TPB, 0, stream>>>(xb, xh, n4x);
    cvt_f16_4<<<cdiv(DFF * 128 / 4, TPB), TPB, 0, stream>>>(L[l].ff1_w, wtmp, DFF * 128 / 4);
    gemm_f16_wmma<<<dim3(ROWS / 64, DFF / 128), TPB, 0, stream>>>(xh, 128, wtmp, L[l].ff1_b, ffhh, DFF, 0, 128, 1, 1);
    cvt_f16_4<<<cdiv(DFF * 128 / 4, TPB), TPB, 0, stream>>>(L[l].ff2_w, wtmp, DFF * 128 / 4);
    gemm_f16_wmma<<<dim3(ROWS / 64, 1), TPB, 0, stream>>>(ffhh, DFF, wtmp, L[l].ff2_b, ob, 128, 0, DFF, 0, 0);
    add_layernorm<<<ROWS / 8, TPB, 0, stream>>>(xb, ob, L[l].ln2_g, L[l].ln2_b);
  }

  // ---- Policy head ----
  head_kernel<<<BN, 128, 0, stream>>>(xb, pa_w1, pa_b1, pa_w2, pa_b2, (float*)d_out);
}
